// GraphAttentionLayer_51762945851834
// MI455X (gfx1250) — matrix operand order repaired and checked by reading, not compiled, not run on hardware
//
#include <hip/hip_runtime.h>
#include <hip/hip_bf16.h>
#include <stdint.h>

#define GAT_ALPHA 0.2f
#define NROWS 8192
#define FIN   256
#define FOUT  64
#define NWAVES 8

typedef __attribute__((ext_vector_type(16))) __bf16         v16bf;
typedef __attribute__((ext_vector_type(8)))  float          v8f;
typedef __attribute__((ext_vector_type(4)))  int            i32x4;
typedef __attribute__((ext_vector_type(4)))  float          f32x4;

__device__ __forceinline__ unsigned short f2bf(float f) {
    // round-to-nearest-even fp32 -> bf16 (used only in the cold pack kernel)
    unsigned int u = __float_as_uint(f);
    unsigned int r = u + 0x7FFFu + ((u >> 16) & 1u);
    return (unsigned short)(r >> 16);
}

// ---------------------------------------------------------------------------
// K1: Wh = h @ W   (8192x256 @ 256x64, fp32, kept full precision)
// ---------------------------------------------------------------------------
__global__ void __launch_bounds__(256) k_wh(const float* __restrict__ h,
                                            const float* __restrict__ W,
                                            float* __restrict__ Wh) {
    int idx = blockIdx.x * 256 + threadIdx.x;   // 0 .. 8192*64-1
    int r = idx >> 6;
    int f = idx & 63;
    const float* hr = h + (size_t)r * FIN;
    float acc = 0.f;
#pragma unroll 8
    for (int k = 0; k < FIN; ++k)
        acc = fmaf(hr[k], W[k * FOUT + f], acc);   // W col reads coalesced over f
    Wh[idx] = acc;
}

// ---------------------------------------------------------------------------
// K2: s1[i] = Wh[i,:].a1 ; s2[i] = Wh[i,:].a2
// ---------------------------------------------------------------------------
__global__ void __launch_bounds__(256) k_s(const float* __restrict__ Wh,
                                           const float* __restrict__ a,
                                           float* __restrict__ s1,
                                           float* __restrict__ s2) {
    int i = blockIdx.x * 256 + threadIdx.x;
    if (i >= NROWS) return;
    const float* wr = Wh + (size_t)i * FOUT;
    float v1 = 0.f, v2 = 0.f;
#pragma unroll
    for (int f = 0; f < FOUT; ++f) {
        float w = wr[f];
        v1 = fmaf(w, a[f],        v1);
        v2 = fmaf(w, a[FOUT + f], v2);
    }
    s1[i] = v1;
    s2[i] = v2;
}

// ---------------------------------------------------------------------------
// K3: maxs2 = max_j s2[j]   (single block reduction)
// ---------------------------------------------------------------------------
__global__ void __launch_bounds__(256) k_max(const float* __restrict__ s2,
                                             float* __restrict__ m) {
    __shared__ float red[256];
    float local = -3.4e38f;
    for (int i = threadIdx.x; i < NROWS; i += 256)
        local = fmaxf(local, s2[i]);
    red[threadIdx.x] = local;
    __syncthreads();
    for (int off = 128; off > 0; off >>= 1) {
        if (threadIdx.x < off)
            red[threadIdx.x] = fmaxf(red[threadIdx.x], red[threadIdx.x + off]);
        __syncthreads();
    }
    if (threadIdx.x == 0) m[0] = red[0];
}

// ---------------------------------------------------------------------------
// K4: pre-pack Wh into WMMA B-operand (32x16 K-major bf16) tiles.
// B layout (16-bit, 32x16): lanes 0-15 hold K=0..7,16..23 (k = 2*vgpr+half, +8 if vgpr>=4),
// lanes 16-31 hold K=8..15,24..31; column N = lane&15. One work-unit writes 16 bf16
// contiguously so the hot kernel loads a v16bf with one coalesced 32B access.
// ---------------------------------------------------------------------------
__global__ void __launch_bounds__(256) k_whb(const float* __restrict__ Wh,
                                             unsigned short* __restrict__ WhB) {
    int gid  = blockIdx.x * 256 + threadIdx.x;  // 0 .. (256 chunks*4 tiles*32 lanes)-1
    int lane = gid & 31;
    int t    = (gid >> 5) & 3;
    int c    = gid >> 7;
    int n    = lane & 15;
    int k0   = (lane >> 4) * 8;                 // 0 or 8
    int j    = c * 32;
    unsigned short* dst = WhB + (size_t)gid * 16;
#pragma unroll
    for (int v = 0; v < 8; ++v) {
        int k = k0 + 2 * v + (v >= 4 ? 8 : 0);
        dst[2 * v]     = f2bf(Wh[(size_t)(j + k)     * FOUT + t * 16 + n]);
        dst[2 * v + 1] = f2bf(Wh[(size_t)(j + k + 1) * FOUT + t * 16 + n]);
    }
}

// ---------------------------------------------------------------------------
// K5: fused masked-softmax + attention@Wh (flash style, one pass over adj).
// 1 block = 16 output rows, NWAVES waves split the j range, WMMA accumulate.
// A layout (16-bit, 16x32): lanes 0-15 row M=lane hold K in {0..7,16..23};
// lanes 16-31 row M=lane-16 hold K in {8..15,24..31}.
// Fully branch-free inner loop: masking is done on the exp() ARGUMENT so the
// compiler never has to guard the transcendental with EXEC branches.
// ---------------------------------------------------------------------------
__global__ void __launch_bounds__(32 * NWAVES) k_attn(
        const int* __restrict__ adj,
        const float* __restrict__ s1,
        const float* __restrict__ s2,
        const float* __restrict__ maxs2p,
        const unsigned short* __restrict__ WhB,
        float* __restrict__ out) {
    __shared__ float accs[NWAVES][4][8][32];   // [wave][tile][vgpr][lane] = 32 KB
    __shared__ float ls[NWAVES][32];

    const int rb     = blockIdx.x;            // 0..511 (16-row block)
    const int wave   = threadIdx.x >> 5;
    const int lane   = threadIdx.x & 31;
    const int laneLo = lane & 15;
    const int hi8    = (lane >> 4) << 3;      // 0 (lanes 0-15) / 8 (lanes 16-31)
    const int row    = rb * 16 + laneLo;

    const float s1r = s1[row];
    const float ms2 = maxs2p[0];
    float xm = s1r + ms2;
    const float mrow = fmaxf(xm, GAT_ALPHA * xm);  // provable per-row max bound

    const int*   arow = adj + (size_t)row * NROWS;
    const v16bf* Bv   = (const v16bf*)WhB;

    v8f acc[4] = {};
    float lsum = 0.f;

    for (int c = wave; c < NROWS / 32; c += NWAVES) {
        const int j = c * 32;
        const int*   ap = arow + j + hi8;
        const float* sp = s2   + j + hi8;

        int   av[16];
        float sv[16];
        // adj is 256MB single-use: stream with non-temporal hint (keep L2 for Wh)
        *(i32x4*)(av +  0) = __builtin_nontemporal_load((const i32x4*)(ap));       // K=hi8+0..3
        *(i32x4*)(av +  4) = __builtin_nontemporal_load((const i32x4*)(ap + 4));   // K=hi8+4..7
        *(i32x4*)(av +  8) = __builtin_nontemporal_load((const i32x4*)(ap + 16));  // K=hi8+16..19
        *(i32x4*)(av + 12) = __builtin_nontemporal_load((const i32x4*)(ap + 20));  // K=hi8+20..23
        *(f32x4*)(sv +  0) = *(const f32x4*)(sp);
        *(f32x4*)(sv +  4) = *(const f32x4*)(sp + 4);
        *(f32x4*)(sv +  8) = *(const f32x4*)(sp + 16);
        *(f32x4*)(sv + 12) = *(const f32x4*)(sp + 20);

        // B tiles: issue early so they overlap the VALU work below
        const v16bf* bp = Bv + (size_t)(c * 4) * 32 + lane;
        v16bf b0 = bp[0];      // N tile 0..15
        v16bf b1 = bp[32];     // N tile 16..31
        v16bf b2 = bp[64];     // N tile 32..47
        v16bf b3 = bp[96];     // N tile 48..63

        v16bf A;
#pragma unroll
        for (int e = 0; e < 16; ++e) {
            float xx  = s1r + sv[e];
            float ee  = fmaxf(xx, GAT_ALPHA * xx);         // leaky_relu (alpha<1)
            float arg = (av[e] > 0) ? (ee - mrow) : -1.0e4f; // masked -> exp underflows to 0
            float p   = __expf(arg);
            lsum += p;
            A[e] = (__bf16)p;                               // hw f32->bf16 (RNE)
        }

        acc[0] = __builtin_amdgcn_wmma_f32_16x16x32_bf16(false, A, false, b0,
                                                         (short)0, acc[0], false, false);
        acc[1] = __builtin_amdgcn_wmma_f32_16x16x32_bf16(false, A, false, b1,
                                                         (short)0, acc[1], false, false);
        acc[2] = __builtin_amdgcn_wmma_f32_16x16x32_bf16(false, A, false, b2,
                                                         (short)0, acc[2], false, false);
        acc[3] = __builtin_amdgcn_wmma_f32_16x16x32_bf16(false, A, false, b3,
                                                         (short)0, acc[3], false, false);
    }

    // lanes L and L+16 hold complementary K halves of row L: combine row sums
    lsum += __shfl_xor(lsum, 16, 32);

#pragma unroll
    for (int t = 0; t < 4; ++t)
#pragma unroll
        for (int g = 0; g < 8; ++g)
            accs[wave][t][g][lane] = acc[t][g];
    ls[wave][lane] = lsum;
    __syncthreads();

    if (wave == 0) {
        float ltot = 0.f;
#pragma unroll
        for (int w = 0; w < NWAVES; ++w) ltot += ls[w][lane];

        float linv[8];
#pragma unroll
        for (int g = 0; g < 8; ++g)
            linv[g] = 1.0f / __shfl(ltot, g + hi8, 32);   // row-sum reciprocal

#pragma unroll
        for (int t = 0; t < 4; ++t) {
#pragma unroll
            for (int g = 0; g < 8; ++g) {
                float v = 0.f;
#pragma unroll
                for (int w = 0; w < NWAVES; ++w) v += accs[w][t][g][lane];
                int   m  = g + hi8;                   // C/D layout: M = vgpr + 8*(lane>=16)
                float hp = v * linv[g];               // softmax normalize
                out[(size_t)(rb * 16 + m) * FOUT + t * 16 + laneLo] =
                    hp > 0.f ? hp : expm1f(hp);       // ELU
            }
        }
    }
}

// ---------------------------------------------------------------------------
// Launch
// inputs: h (8192x256 f32), adj (8192x8192 i32), W (256x64 f32), a (128x1 f32)
// out:    8192x64 f32
// ---------------------------------------------------------------------------
extern "C" void kernel_launch(void* const* d_in, const int* in_sizes, int n_in,
                              void* d_out, int out_size, void* d_ws, size_t ws_size,
                              hipStream_t stream) {
    const float* h   = (const float*)d_in[0];
    const int*   adj = (const int*)d_in[1];
    const float* W   = (const float*)d_in[2];
    const float* a   = (const float*)d_in[3];
    float* out = (float*)d_out;

    char* ws = (char*)d_ws;
    float* Wh  = (float*)(ws);                                 // 2 MB
    float* s1  = (float*)(ws + (size_t)2097152);               // 32 KB
    float* s2  = (float*)(ws + (size_t)2097152 + 32768);       // 32 KB
    float* ms2 = (float*)(ws + (size_t)2097152 + 65536);       // 4 B (+pad)
    unsigned short* WhB = (unsigned short*)(ws + (size_t)2097152 + 65536 + 64); // 1 MB, 32B-aligned

    k_wh  <<<dim3((NROWS * FOUT) / 256), dim3(256), 0, stream>>>(h, W, Wh);
    k_s   <<<dim3(NROWS / 256),          dim3(256), 0, stream>>>(Wh, a, s1, s2);
    k_max <<<dim3(1),                    dim3(256), 0, stream>>>(s2, ms2);
    k_whb <<<dim3((NROWS / 32) * 4 * 32 / 256), dim3(256), 0, stream>>>(Wh, WhB);
    k_attn<<<dim3(NROWS / 16),   dim3(32 * NWAVES), 0, stream>>>(adj, s1, s2, ms2, WhB, out);
}
